// MultiHeadAttention_39152921870896
// MI455X (gfx1250) — matrix operand order repaired and checked by reading, not compiled, not run on hardware
//
#include <hip/hip_runtime.h>
#include <hip/hip_bf16.h>

#define D_MODEL 1024
#define SEQ     2048
#define BATCH   2
#define HEADS   16
#define DKH     64
#define BS      (BATCH * SEQ)   // 4096 rows

typedef __attribute__((ext_vector_type(16))) _Float16 v16h;
typedef __attribute__((ext_vector_type(8)))  _Float16 v8h;
typedef __attribute__((ext_vector_type(4)))  _Float16 v4h;
typedef __attribute__((ext_vector_type(8)))  float    v8f;

#define WMMA_F16(a, b, c) \
  __builtin_amdgcn_wmma_f32_16x16x32_f16(false, (a), false, (b), (short)0, (c), false, false)

// ---------------- WMMA fragment loaders (wave32, ISA 7.12.2 layouts) ------------
// A (16x32 f16): lane L<16 holds row M=L, K = {0..7, 16..23}; lane L+16: K = {8..15, 24..31}.
// v16h element 2j/2j+1 == low/high half of VGPR j.

static __device__ __forceinline__ v16h load_A_f16(const _Float16* __restrict__ A, int ld) {
  const int lane = threadIdx.x & 31;
  const _Float16* row = A + (size_t)(lane & 15) * ld + ((lane >> 4) << 3);
  v8h lo = *(const v8h*)row;
  v8h hi = *(const v8h*)(row + 16);
  return __builtin_shufflevector(lo, hi, 0, 1, 2, 3, 4, 5, 6, 7, 8, 9, 10, 11, 12, 13, 14, 15);
}

static __device__ __forceinline__ v16h load_A_lds_f32(const float* P, int ld) {
  const int lane = threadIdx.x & 31;
  const float* row = P + (lane & 15) * ld + ((lane >> 4) << 3);
  v16h a;
#pragma unroll
  for (int j = 0; j < 8; ++j) {
    a[j]     = (_Float16)row[j];
    a[8 + j] = (_Float16)row[16 + j];
  }
  return a;
}

// B (32x16): lane holds column N = lane&15; lanes 0-15 cover K = {0..7, 16..23}, lanes 16-31 K = {8..15, 24..31}.
// source is row-major M[n][k]: each B column = one contiguous source row (W^T / K^T / V^T cases).
static __device__ __forceinline__ v16h load_B_rows_f16(const _Float16* __restrict__ Br, int ld) {
  const int lane = threadIdx.x & 31;
  const _Float16* src = Br + (size_t)(lane & 15) * ld + ((lane >> 4) << 3);
  v8h lo = *(const v8h*)src;
  v8h hi = *(const v8h*)(src + 16);
  return __builtin_shufflevector(lo, hi, 0, 1, 2, 3, 4, 5, 6, 7, 8, 9, 10, 11, 12, 13, 14, 15);
}

// ---------------- Kernel 0: elementwise f32 -> f16 conversion -------------------
__global__ __launch_bounds__(256) void cvt_kernel(const float* __restrict__ src,
                                                  _Float16* __restrict__ dst, int n4) {
  const int i = (blockIdx.x * 256 + threadIdx.x) << 2;
  if (i < n4 * 4) {
    const float4 f = *(const float4*)(src + i);
    v4h h;
    h[0] = (_Float16)f.x; h[1] = (_Float16)f.y;
    h[2] = (_Float16)f.z; h[3] = (_Float16)f.w;
    *(v4h*)(dst + i) = h;
  }
}

// ---------------- Kernel 1: projection, f16 x f16 GEMM, 32x32 tile per wave -----
// Y = X @ W^T + b.  vtrans==0: Out[b,h,s,dk] (Q/K).  vtrans!=0: Out[b,h,dk,s] (V).
__global__ __launch_bounds__(256) void proj16_kernel(const _Float16* __restrict__ X,
                                                     const _Float16* __restrict__ W,
                                                     const float* __restrict__ bias,
                                                     _Float16* __restrict__ Out,
                                                     int vtrans) {
  const int wv = threadIdx.x >> 5;
  const int m0 = (blockIdx.y * 8 + wv) * 32;   // 2 rowgroups of 16
  const int n0 = blockIdx.x * 32;              // 2 col tiles of 16
  v8f acc00 = {}, acc01 = {}, acc10 = {}, acc11 = {};
#pragma unroll 2
  for (int kk = 0; kk < D_MODEL; kk += 32) {
    // prefetch next k-step (global_prefetch_b8)
    __builtin_prefetch(X + (size_t)m0 * D_MODEL + kk + 64, 0, 1);
    __builtin_prefetch(W + (size_t)n0 * D_MODEL + kk + 64, 0, 1);
    v16h a0 = load_A_f16(X + (size_t)m0 * D_MODEL + kk, D_MODEL);
    v16h a1 = load_A_f16(X + (size_t)(m0 + 16) * D_MODEL + kk, D_MODEL);
    v16h b0 = load_B_rows_f16(W + (size_t)n0 * D_MODEL + kk, D_MODEL);
    v16h b1 = load_B_rows_f16(W + (size_t)(n0 + 16) * D_MODEL + kk, D_MODEL);
    acc00 = WMMA_F16(a0, b0, acc00);
    acc01 = WMMA_F16(a0, b1, acc01);
    acc10 = WMMA_F16(a1, b0, acc10);
    acc11 = WMMA_F16(a1, b1, acc11);
  }
  const int lane = threadIdx.x & 31;
  const int cn = lane & 15;
  const int mb = (lane >> 4) << 3;
  v8f accs[2][2] = {{acc00, acc01}, {acc10, acc11}};
#pragma unroll
  for (int rg = 0; rg < 2; ++rg) {
#pragma unroll
    for (int ntl = 0; ntl < 2; ++ntl) {
      const int n = n0 + ntl * 16 + cn;
      const float bv = bias[n];
      const int h = n >> 6, dk = n & (DKH - 1);
      const int mrow = m0 + rg * 16 + mb;
      const int bb = mrow >> 11;               // 8-row run never crosses batch
      const int s0 = mrow & (SEQ - 1);
      if (vtrans) {
        // [B,H,DKH,SEQ]: 8 consecutive s per lane -> packed 16B store
        v8h pack;
#pragma unroll
        for (int i = 0; i < 8; ++i) pack[i] = (_Float16)(accs[rg][ntl][i] + bv);
        *(v8h*)(Out + (((size_t)(bb * HEADS + h)) * DKH + dk) * SEQ + s0) = pack;
      } else {
        // [B,H,SEQ,DKH]
#pragma unroll
        for (int i = 0; i < 8; ++i)
          Out[(((size_t)(bb * HEADS + h)) * SEQ + (s0 + i)) * DKH + dk] =
              (_Float16)(accs[rg][ntl][i] + bv);
      }
    }
  }
}

// ---------------- Kernel 2: fused attention per (b,h,32-query-row block) --------
// dynamic LDS: 32 x SEQ f32 scores (256 KB, fits 320 KB/WGP)
__global__ __launch_bounds__(256) void attn_kernel(const _Float16* __restrict__ Qh,
                                                   const _Float16* __restrict__ Kh,
                                                   const _Float16* __restrict__ Vt,
                                                   float* __restrict__ Attn,
                                                   _Float16* __restrict__ Ctx) {
  extern __shared__ float sP[];                 // [32][SEQ]
  const int qblk = blockIdx.x & 63;             // SEQ/32 = 64 query blocks
  const int bh   = blockIdx.x >> 6;             // b*H + h
  const int q0   = qblk * 32;
  const size_t hb = (size_t)bh * SEQ * DKH;
  const int wv   = threadIdx.x >> 5;
  const int lane = threadIdx.x & 31;
  const int cn   = lane & 15;
  const int mb   = (lane >> 4) << 3;

  // ---- Phase 1: scores = (Q K^T) * 1/sqrt(dk) into LDS ----
  {
    const int rg = wv & 1;                      // each wave owns one rowgroup parity
    // Q fragments are loop-invariant: hoist (d_k = 64 -> two k-steps)
    const v16h a0 = load_A_f16(Qh + hb + (size_t)(q0 + rg * 16) * DKH + 0,  DKH);
    const v16h a1 = load_A_f16(Qh + hb + (size_t)(q0 + rg * 16) * DKH + 32, DKH);
    for (int kt = (wv >> 1); kt < SEQ / 16; kt += 4) {
      // prefetch this wave's next K tile (64 rows ahead)
      __builtin_prefetch(Kh + hb + (size_t)((kt + 4) * 16) * DKH, 0, 1);
      v8f acc = {};
      v16h b0 = load_B_rows_f16(Kh + hb + (size_t)(kt * 16) * DKH + 0,  DKH);
      v16h b1 = load_B_rows_f16(Kh + hb + (size_t)(kt * 16) * DKH + 32, DKH);
      acc = WMMA_F16(a0, b0, acc);
      acc = WMMA_F16(a1, b1, acc);
      float* dst = sP + (rg * 16 + mb) * SEQ + kt * 16 + cn;
#pragma unroll
      for (int i = 0; i < 8; ++i) dst[(size_t)i * SEQ] = acc[i] * 0.125f;  // 1/sqrt(64)
    }
  }
  __syncthreads();

  // ---- Phase 2: f32 softmax over each of 32 rows; write attn to d_out ----
  {
    const int row = threadIdx.x >> 3;           // 8 threads per row
    const int tin = threadIdx.x & 7;
    float* pr = sP + row * SEQ;
    float mx = -3.4e38f;
    for (int j = tin; j < SEQ; j += 8) mx = fmaxf(mx, pr[j]);
#pragma unroll
    for (int off = 4; off; off >>= 1) mx = fmaxf(mx, __shfl_xor(mx, off, 8));
    float sum = 0.f;
    for (int j = tin; j < SEQ; j += 8) {        // exp once, stored in place
      const float e = __expf(pr[j] - mx);
      sum += e;
      pr[j] = e;
    }
#pragma unroll
    for (int off = 4; off; off >>= 1) sum += __shfl_xor(sum, off, 8);
    const float inv = 1.0f / sum;
    float* arow = Attn + ((size_t)bh * SEQ + (q0 + row)) * SEQ;
    for (int j = tin; j < SEQ; j += 8) {
      const float p = pr[j] * inv;
      pr[j]   = p;                              // keep for P@V
      arow[j] = p;                              // attention output
    }
  }
  __syncthreads();

  // ---- Phase 3: ctx = P @ V, P from LDS (f32->f16), V transposed [B,H,DKH,SEQ] ----
  {
    const int rg = wv >> 2;                     // 2 rowgroups x 4 n-tiles = 8 waves
    const int nt = wv & 3;
    v8f acc = {};
    for (int kc = 0; kc < SEQ; kc += 32) {
      v16h a = load_A_lds_f32(sP + (rg * 16) * SEQ + kc, SEQ);
      // B[k][n] = V[s=kc+k][dk=nt*16+n] = Vt[nt*16+n][kc+k] : contiguous rows, ld=SEQ
      v16h b = load_B_rows_f16(Vt + hb + (size_t)(nt * 16) * SEQ + kc, SEQ);
      acc = WMMA_F16(a, b, acc);
    }
    const int hh = bh & (HEADS - 1);
    const int bb = bh >> 4;
#pragma unroll
    for (int i = 0; i < 8; ++i) {
      const int s = q0 + rg * 16 + mb + i;
      Ctx[((size_t)bb * SEQ + s) * D_MODEL + hh * DKH + nt * 16 + cn] = (_Float16)acc[i];
    }
  }
}

// ---------------- Kernel 3: out = ctx_f16 @ Wo16^T + bo -> f32, 32x32/wave -------
__global__ __launch_bounds__(256) void out_proj16_kernel(const _Float16* __restrict__ Xh,
                                                         const _Float16* __restrict__ W,
                                                         const float* __restrict__ bias,
                                                         float* __restrict__ Out) {
  const int wv = threadIdx.x >> 5;
  const int m0 = (blockIdx.y * 8 + wv) * 32;
  const int n0 = blockIdx.x * 32;
  v8f acc00 = {}, acc01 = {}, acc10 = {}, acc11 = {};
#pragma unroll 2
  for (int kk = 0; kk < D_MODEL; kk += 32) {
    __builtin_prefetch(Xh + (size_t)m0 * D_MODEL + kk + 64, 0, 1);
    v16h a0 = load_A_f16(Xh + (size_t)m0 * D_MODEL + kk, D_MODEL);
    v16h a1 = load_A_f16(Xh + (size_t)(m0 + 16) * D_MODEL + kk, D_MODEL);
    v16h b0 = load_B_rows_f16(W + (size_t)n0 * D_MODEL + kk, D_MODEL);
    v16h b1 = load_B_rows_f16(W + (size_t)(n0 + 16) * D_MODEL + kk, D_MODEL);
    acc00 = WMMA_F16(a0, b0, acc00);
    acc01 = WMMA_F16(a0, b1, acc01);
    acc10 = WMMA_F16(a1, b0, acc10);
    acc11 = WMMA_F16(a1, b1, acc11);
  }
  const int lane = threadIdx.x & 31;
  const int cn = lane & 15;
  const int mb = (lane >> 4) << 3;
  v8f accs[2][2] = {{acc00, acc01}, {acc10, acc11}};
#pragma unroll
  for (int rg = 0; rg < 2; ++rg) {
#pragma unroll
    for (int ntl = 0; ntl < 2; ++ntl) {
      const int n = n0 + ntl * 16 + cn;
      const float bv = bias[n];
#pragma unroll
      for (int i = 0; i < 8; ++i) {
        const int m = m0 + rg * 16 + mb + i;
        Out[(size_t)m * D_MODEL + n] = accs[rg][ntl][i] + bv;
      }
    }
  }
}

// ---------------- launch -------------------------------------------------------
extern "C" void kernel_launch(void* const* d_in, const int* in_sizes, int n_in,
                              void* d_out, int out_size, void* d_ws, size_t ws_size,
                              hipStream_t stream) {
  const float* q  = (const float*)d_in[0];
  const float* k  = (const float*)d_in[1];
  const float* v  = (const float*)d_in[2];
  const float* Wq = (const float*)d_in[3];
  const float* bq = (const float*)d_in[4];
  const float* Wk = (const float*)d_in[5];
  const float* bk = (const float*)d_in[6];
  const float* Wv = (const float*)d_in[7];
  const float* bv = (const float*)d_in[8];
  const float* Wo = (const float*)d_in[9];
  const float* bo = (const float*)d_in[10];

  float* out  = (float*)d_out;                         // [B,S,D] f32
  float* attn = out + (size_t)BS * D_MODEL;            // [B,H,S,S] f32

  // workspace layout (all f16):
  //  q16,k16,v16    : 3 x BS*D        (24 MB)
  //  Wq16..Wo16     : 4 x D*D         ( 8 MB)
  //  Qh, Kh         : 2 x [B,H,S,dk]  (16 MB)
  //  Vt             : [B,H,dk,S]      ( 8 MB)
  //  Ctx            : [B,S,D]         ( 8 MB)
  char* ws = (char*)d_ws;
  const size_t xBytes = (size_t)BS * D_MODEL * sizeof(_Float16);       // 8 MB
  const size_t wBytes = (size_t)D_MODEL * D_MODEL * sizeof(_Float16);  // 2 MB
  _Float16* q16  = (_Float16*)(ws);
  _Float16* k16  = (_Float16*)(ws + xBytes);
  _Float16* v16  = (_Float16*)(ws + 2 * xBytes);
  _Float16* Wq16 = (_Float16*)(ws + 3 * xBytes);
  _Float16* Wk16 = (_Float16*)(ws + 3 * xBytes + wBytes);
  _Float16* Wv16 = (_Float16*)(ws + 3 * xBytes + 2 * wBytes);
  _Float16* Wo16 = (_Float16*)(ws + 3 * xBytes + 3 * wBytes);
  _Float16* Qh   = (_Float16*)(ws + 3 * xBytes + 4 * wBytes);
  _Float16* Kh   = (_Float16*)(ws + 4 * xBytes + 4 * wBytes);
  _Float16* Vt   = (_Float16*)(ws + 5 * xBytes + 4 * wBytes);
  _Float16* Ctx  = (_Float16*)(ws + 6 * xBytes + 4 * wBytes);

  // 0) one-time f32 -> f16 conversions
  const int nX4 = BS * D_MODEL / 4, nW4 = D_MODEL * D_MODEL / 4;
  cvt_kernel<<<nX4 / 256, 256, 0, stream>>>(q,  q16,  nX4);
  cvt_kernel<<<nX4 / 256, 256, 0, stream>>>(k,  k16,  nX4);
  cvt_kernel<<<nX4 / 256, 256, 0, stream>>>(v,  v16,  nX4);
  cvt_kernel<<<nW4 / 256, 256, 0, stream>>>(Wq, Wq16, nW4);
  cvt_kernel<<<nW4 / 256, 256, 0, stream>>>(Wk, Wk16, nW4);
  cvt_kernel<<<nW4 / 256, 256, 0, stream>>>(Wv, Wv16, nW4);
  cvt_kernel<<<nW4 / 256, 256, 0, stream>>>(Wo, Wo16, nW4);

  // 1) projections (pure f16 WMMA GEMMs), 32x32 tile per wave
  const dim3 gproj(D_MODEL / 32, BS / (8 * 32));       // (32, 16)
  proj16_kernel<<<gproj, 256, 0, stream>>>(q16, Wq16, bq, Qh, 0);
  proj16_kernel<<<gproj, 256, 0, stream>>>(k16, Wk16, bk, Kh, 0);
  proj16_kernel<<<gproj, 256, 0, stream>>>(v16, Wv16, bv, Vt, 1);  // transposed V

  // 2) fused attention (scores -> softmax -> attn out -> P@V), 256 KB dynamic LDS
  const size_t ldsBytes = (size_t)32 * SEQ * sizeof(float);
  attn_kernel<<<dim3(BATCH * HEADS * (SEQ / 32)), 256, ldsBytes, stream>>>(Qh, Kh, Vt, attn, Ctx);

  // 3) output projection
  out_proj16_kernel<<<gproj, 256, 0, stream>>>(Ctx, Wo16, bo, out);
}
